// MultiHeadAttention_25151328485592
// MI455X (gfx1250) — matrix operand order repaired and checked by reading, not compiled, not run on hardware
//
#include <hip/hip_runtime.h>
#include <cstdint>
#include <cstddef>

typedef __attribute__((ext_vector_type(16))) _Float16 v16h;
typedef __attribute__((ext_vector_type(8)))  _Float16 v8h;
typedef __attribute__((ext_vector_type(8)))  float    v8f;

#define D_MODEL 1024
#define N_HEADS 16
#define DK      64
#define SEQ     2048
#define BATCH   2
#define M_TOTAL (BATCH * SEQ)   // 4096 rows

union AFrag {
    v16h v;
    v8h  h[2];
    _Float16 e[16];
};

// ---------------------------------------------------------------------------
// Stage 0: fp32 -> f16 conversion
// ---------------------------------------------------------------------------
__global__ void cvt_f32_f16(const float* __restrict__ in,
                            _Float16* __restrict__ out, int n) {
    int i = blockIdx.x * blockDim.x + threadIdx.x;
    if (i < n) out[i] = (_Float16)in[i];
}

// ---------------------------------------------------------------------------
// Stage 1/3: Y[m,n] = sum_k X[m,k] * W[n,k] + bias[n], K = D_MODEL (static).
// One wave -> 16x64 output strip: 1 A-frag feeds 4 WMMAs per K-step.
// Block = 128 threads = 4 waves covering a 64x64 output block.
// ---------------------------------------------------------------------------
template <typename OutT>
__global__ void __launch_bounds__(128, 1)
gemm_xwt(const _Float16* __restrict__ X,
         const _Float16* __restrict__ W,
         const float* __restrict__ bias,
         OutT* __restrict__ Y,
         int M, int N) {
    constexpr int K = D_MODEL;
    const int lane = threadIdx.x & 31;
    const int wave = threadIdx.x >> 5;
    const int g  = lane >> 4;     // half-wave group
    const int lm = lane & 15;
    const int m0 = (blockIdx.y * 4 + wave) * 16;
    const int n0 = blockIdx.x * 64;

    const _Float16* arow = X + (size_t)(m0 + lm) * K;  // A: lane = M row
    const _Float16* br0  = W + (size_t)(n0 +      lm) * K;
    const _Float16* br1  = W + (size_t)(n0 + 16 + lm) * K;
    const _Float16* br2  = W + (size_t)(n0 + 32 + lm) * K;
    const _Float16* br3  = W + (size_t)(n0 + 48 + lm) * K;

    v8f acc[4] = {{}, {}, {}, {}};
#pragma unroll 4
    for (int k0 = 0; k0 < K; k0 += 32) {
        AFrag a;
        // A, contiguous k order (same as B): lanes 0-15 hold K = k0+0..15; lanes 16-31 K = k0+16..31
        a.h[0] = *(const v8h*)(arow + k0 + g * 16);
        a.h[1] = *(const v8h*)(arow + k0 + g * 16 + 8);
        // B, contiguous k order (same as A): lanes 0-15 K = k0+0..15; lanes 16-31 K = k0+16..31
        AFrag b0, b1, b2, b3;
        b0.v = *(const v16h*)(br0 + k0 + g * 16);
        b1.v = *(const v16h*)(br1 + k0 + g * 16);
        b2.v = *(const v16h*)(br2 + k0 + g * 16);
        b3.v = *(const v16h*)(br3 + k0 + g * 16);
        acc[0] = __builtin_amdgcn_wmma_f32_16x16x32_f16(false, a.v, false, b0.v, (short)0, acc[0], false, false);
        acc[1] = __builtin_amdgcn_wmma_f32_16x16x32_f16(false, a.v, false, b1.v, (short)0, acc[1], false, false);
        acc[2] = __builtin_amdgcn_wmma_f32_16x16x32_f16(false, a.v, false, b2.v, (short)0, acc[2], false, false);
        acc[3] = __builtin_amdgcn_wmma_f32_16x16x32_f16(false, a.v, false, b3.v, (short)0, acc[3], false, false);
    }
#pragma unroll
    for (int t = 0; t < 4; ++t) {
        const float bn = bias ? bias[n0 + t * 16 + lm] : 0.0f;
#pragma unroll
        for (int r = 0; r < 8; ++r) {
            // D layout: M = r + 8*g, N = lm
            const int m = m0 + r + 8 * g;
            Y[(size_t)m * N + n0 + t * 16 + lm] = (OutT)(acc[t][r] + bn);
        }
    }
}

// ---------------------------------------------------------------------------
// Stage 2: causal flash attention.
// One wave per (batch, head, 16-row q tile); 32 keys per iteration.
// V block staged via async global->LDS (ASYNCcnt) copies.
// ---------------------------------------------------------------------------
__global__ void __launch_bounds__(32, 1)
attention_kernel(const _Float16* __restrict__ Q,
                 const _Float16* __restrict__ Kx,
                 const _Float16* __restrict__ V,
                 _Float16* __restrict__ Ctx) {
    __shared__ __align__(32) _Float16 Pld[16 * 32];   // P tile: 16 q x 32 k
    __shared__ __align__(32) _Float16 Vld[32 * DK];   // V block: 32 k x 64 d

    const int lane = threadIdx.x & 31;
    const int g  = lane >> 4;
    const int lm = lane & 15;
    const int q0 = blockIdx.x * 16;
    const int h  = blockIdx.y;
    const int b  = blockIdx.z;

    const size_t base = (size_t)b * SEQ * D_MODEL + (size_t)h * DK;
    const _Float16* Qb = Q  + base;
    const _Float16* Kb = Kx + base;
    const _Float16* Vb = V  + base;

    // Q A-fragments for d = 0..31 (a0) and d = 32..63 (a1)
    AFrag a0, a1;
    {
        const _Float16* qrow = Qb + (size_t)(q0 + lm) * D_MODEL;
        a0.h[0] = *(const v8h*)(qrow + g * 16);
        a0.h[1] = *(const v8h*)(qrow + g * 16 + 8);
        a1.h[0] = *(const v8h*)(qrow + 32 + g * 16);
        a1.h[1] = *(const v8h*)(qrow + 32 + g * 16 + 8);
    }

    v8f o[4] = {{}, {}, {}, {}};      // context accum, 4 dk tiles of 16
    float mrow[8], lrow[8];
#pragma unroll
    for (int r = 0; r < 8; ++r) { mrow[r] = -1e30f; lrow[r] = 0.0f; }

    const float scale = 0.125f;       // 1/sqrt(64)
    const int jend = q0 + 16;         // causal: keys <= q0+15
    const unsigned vlds = (unsigned)(uintptr_t)&Vld[lane * DK];  // LDS byte offset

    for (int j0 = 0; j0 < jend; j0 += 32) {
        // ---- async-stage V block (rows j0..j0+31, 128B each) into LDS
        {
            const _Float16* vrow = Vb + (size_t)(j0 + lane) * D_MODEL;
            asm volatile(
                "global_load_async_to_lds_b128 %0, %1, off\n\t"
                "global_load_async_to_lds_b128 %0, %1, off offset:16\n\t"
                "global_load_async_to_lds_b128 %0, %1, off offset:32\n\t"
                "global_load_async_to_lds_b128 %0, %1, off offset:48\n\t"
                "global_load_async_to_lds_b128 %0, %1, off offset:64\n\t"
                "global_load_async_to_lds_b128 %0, %1, off offset:80\n\t"
                "global_load_async_to_lds_b128 %0, %1, off offset:96\n\t"
                "global_load_async_to_lds_b128 %0, %1, off offset:112"
                :: "v"(vlds), "v"(vrow) : "memory");
        }
        if (j0 + 32 < jend)           // gfx1250 global_prefetch_b8
            __builtin_prefetch(Kb + (size_t)(j0 + 32 + lane) * D_MODEL, 0, 0);

        // ---- S = Q*K^T for 32 keys: two 16x16 tiles, d-dim 64 = 2 WMMAs each
        v8f s0 = {}, s1 = {};
        {
            const _Float16* k0r = Kb + (size_t)(j0 + lm) * D_MODEL;
            AFrag bk0, bk1;
            bk0.v = *(const v16h*)(k0r + g * 16);
            bk1.v = *(const v16h*)(k0r + 32 + g * 16);
            s0 = __builtin_amdgcn_wmma_f32_16x16x32_f16(false, a0.v, false, bk0.v, (short)0, s0, false, false);
            s0 = __builtin_amdgcn_wmma_f32_16x16x32_f16(false, a1.v, false, bk1.v, (short)0, s0, false, false);
        }
        {
            const _Float16* k1r = Kb + (size_t)(j0 + 16 + lm) * D_MODEL;
            AFrag bk0, bk1;
            bk0.v = *(const v16h*)(k1r + g * 16);
            bk1.v = *(const v16h*)(k1r + 32 + g * 16);
            s1 = __builtin_amdgcn_wmma_f32_16x16x32_f16(false, a0.v, false, bk0.v, (short)0, s1, false, false);
            s1 = __builtin_amdgcn_wmma_f32_16x16x32_f16(false, a1.v, false, bk1.v, (short)0, s1, false, false);
        }

        // ---- online softmax (fp32), write P tile (f16) to LDS
        const bool edge = (j0 + 31 >= q0);   // wave-uniform: masking needed?
#pragma unroll
        for (int r = 0; r < 8; ++r) {
            const int qi = q0 + r + 8 * g;
            float v0 = s0[r] * scale;
            float v1 = s1[r] * scale;
            if (edge) {
                if (j0 + lm      > qi) v0 = -1e30f;
                if (j0 + 16 + lm > qi) v1 = -1e30f;
            }
            float mx = fmaxf(v0, v1);
            mx = fmaxf(mx, __shfl_xor(mx, 1, 32));
            mx = fmaxf(mx, __shfl_xor(mx, 2, 32));
            mx = fmaxf(mx, __shfl_xor(mx, 4, 32));
            mx = fmaxf(mx, __shfl_xor(mx, 8, 32));

            const float nm = fmaxf(mrow[r], mx);
            const float al = __expf(mrow[r] - nm);
            mrow[r] = nm;

            const float e0 = __expf(v0 - nm);
            const float e1 = __expf(v1 - nm);
            float sum = e0 + e1;
            sum += __shfl_xor(sum, 1, 32);
            sum += __shfl_xor(sum, 2, 32);
            sum += __shfl_xor(sum, 4, 32);
            sum += __shfl_xor(sum, 8, 32);
            lrow[r] = al * lrow[r] + sum;

            Pld[(r + 8 * g) * 32 + lm]      = (_Float16)e0;
            Pld[(r + 8 * g) * 32 + 16 + lm] = (_Float16)e1;

#pragma unroll
            for (int t = 0; t < 4; ++t) o[t][r] *= al;
        }
        __syncthreads();
        asm volatile("s_wait_asynccnt 0x0" ::: "memory");   // V block visible in LDS

        // ---- P (A-frag, M=16 q, K=32 keys) from LDS
        AFrag pa;
        pa.h[0] = *(const v8h*)(&Pld[lm * 32 + g * 16]);
        pa.h[1] = *(const v8h*)(&Pld[lm * 32 + g * 16 + 8]);

        // ---- O += P * V : 4 dk tiles of 16, B-frags gathered from LDS
#pragma unroll
        for (int t = 0; t < 4; ++t) {
            AFrag vb;
#pragma unroll
            for (int i = 0; i < 16; ++i)
                vb.e[i] = Vld[(g * 16 + i) * DK + t * 16 + lm];
            o[t] = __builtin_amdgcn_wmma_f32_16x16x32_f16(
                       false, pa.v, false, vb.v, (short)0, o[t], false, false);
        }
        __syncthreads();
    }

    // ---- normalize and store context (f16, [b,s,e] layout)
#pragma unroll
    for (int r = 0; r < 8; ++r) {
        const float inv = 1.0f / lrow[r];
        const int m = q0 + r + 8 * g;
        _Float16* crow = Ctx + ((size_t)b * SEQ + m) * D_MODEL + h * DK;
#pragma unroll
        for (int t = 0; t < 4; ++t)
            crow[t * 16 + lm] = (_Float16)(o[t][r] * inv);
    }
}

// ---------------------------------------------------------------------------
// Host-side orchestration
// ---------------------------------------------------------------------------
extern "C" void kernel_launch(void* const* d_in, const int* in_sizes, int n_in,
                              void* d_out, int out_size, void* d_ws, size_t ws_size,
                              hipStream_t stream) {
    const float* x  = (const float*)d_in[0];
    const float* wq = (const float*)d_in[1];
    const float* bq = (const float*)d_in[2];
    const float* wk = (const float*)d_in[3];
    const float* bk = (const float*)d_in[4];
    const float* wv = (const float*)d_in[5];
    const float* bv = (const float*)d_in[6];
    const float* wo = (const float*)d_in[7];
    const float* bo = (const float*)d_in[8];
    float* out = (float*)d_out;

    const size_t XN = (size_t)M_TOTAL * D_MODEL;   // 4,194,304
    const size_t WN = (size_t)D_MODEL * D_MODEL;   // 1,048,576

    _Float16* Xh   = (_Float16*)d_ws;
    _Float16* Wqh  = Xh  + XN;
    _Float16* Wkh  = Wqh + WN;
    _Float16* Wvh  = Wkh + WN;
    _Float16* Woh  = Wvh + WN;
    _Float16* Qh   = Woh + WN;
    _Float16* Kh   = Qh  + XN;
    _Float16* Vh   = Kh  + XN;
    _Float16* Ctxh = Vh  + XN;        // total ~48 MB of f16 scratch

    // Stage 0: conversions
    cvt_f32_f16<<<(int)((XN + 255) / 256), 256, 0, stream>>>(x,  Xh,  (int)XN);
    cvt_f32_f16<<<(int)((WN + 255) / 256), 256, 0, stream>>>(wq, Wqh, (int)WN);
    cvt_f32_f16<<<(int)((WN + 255) / 256), 256, 0, stream>>>(wk, Wkh, (int)WN);
    cvt_f32_f16<<<(int)((WN + 255) / 256), 256, 0, stream>>>(wv, Wvh, (int)WN);
    cvt_f32_f16<<<(int)((WN + 255) / 256), 256, 0, stream>>>(wo, Woh, (int)WN);

    // Stage 1: Q/K/V projections (WMMA GEMM, f16 out)
    dim3 ggrid(D_MODEL / 64, M_TOTAL / 64);   // (16, 64)
    dim3 gblk(128);
    gemm_xwt<_Float16><<<ggrid, gblk, 0, stream>>>(Xh, Wqh, bq, Qh, M_TOTAL, D_MODEL);
    gemm_xwt<_Float16><<<ggrid, gblk, 0, stream>>>(Xh, Wkh, bk, Kh, M_TOTAL, D_MODEL);
    gemm_xwt<_Float16><<<ggrid, gblk, 0, stream>>>(Xh, Wvh, bv, Vh, M_TOTAL, D_MODEL);

    // Stage 2: causal flash attention
    dim3 agrid(SEQ / 16, N_HEADS, BATCH);     // (128, 16, 2)
    attention_kernel<<<agrid, 32, 0, stream>>>(Qh, Kh, Vh, Ctxh);

    // Stage 3: output projection (fp32 out)
    gemm_xwt<float><<<ggrid, gblk, 0, stream>>>(Ctxh, Woh, bo, out, M_TOTAL, D_MODEL);
}
